// TopKSAE_44160853737877
// MI455X (gfx1250) — matrix operand order repaired and checked by reading, not compiled, not run on hardware
//
#include <hip/hip_runtime.h>

#define DIM    768
#define HIDDEN 24576
#define BATCH  8192
#define TOPK   32

typedef __attribute__((ext_vector_type(16))) __bf16 v16bf;
typedef __attribute__((ext_vector_type(8)))  float  v8f;

__device__ __forceinline__ unsigned short f2bf(float f) {
    unsigned u = __builtin_bit_cast(unsigned, f);
    u += 0x7FFFu + ((u >> 16) & 1u);          // round-to-nearest-even
    return (unsigned short)(u >> 16);
}
__device__ __forceinline__ float bf2f(unsigned short h) {
    unsigned u = ((unsigned)h) << 16;
    return __builtin_bit_cast(float, u);
}

// LDS byte offset = low 32 bits of the shared-aperture flat address (ISA §10.2)
__device__ __forceinline__ unsigned ldsOff(const void* p) {
    return (unsigned)(unsigned long long)p;
}

// CDNA5 async global->LDS copy, 16 bytes per lane, tracked by ASYNCcnt
__device__ __forceinline__ void asyncCp16(unsigned ldsByte, unsigned gByte, const void* base) {
    asm volatile("global_load_async_to_lds_b128 %0, %1, %2"
                 :: "v"(ldsByte), "v"(gByte), "s"((unsigned long long)base)
                 : "memory");
}

// ---------------------------------------------------------------------------
// Prep: W_enc f32 -> bf16 (L2-resident working copy)
// ---------------------------------------------------------------------------
__global__ void conv_wenc(const float* __restrict__ W, unsigned short* __restrict__ Wb) {
    const int n4 = (HIDDEN * DIM) / 4;
    for (int i = blockIdx.x * blockDim.x + threadIdx.x; i < n4; i += gridDim.x * blockDim.x) {
        float4 f = ((const float4*)W)[i];
        ushort4 o;
        o.x = f2bf(f.x); o.y = f2bf(f.y); o.z = f2bf(f.z); o.w = f2bf(f.w);
        ((ushort4*)Wb)[i] = o;
    }
}

// Prep: xc = x - b_dec, f32 -> bf16
__global__ void conv_x(const float* __restrict__ x, const float* __restrict__ bdec,
                       unsigned short* __restrict__ xb) {
    const int b = blockIdx.x;
    for (int d = threadIdx.x; d < DIM; d += blockDim.x)
        xb[(size_t)b * DIM + d] = f2bf(x[(size_t)b * DIM + d] - bdec[d]);
}

// ---------------------------------------------------------------------------
// Encoder GEMM: pre[b,h] = xc[b,:] . W_enc[h,:] + b_enc[h]   (bf16 WMMA)
// Block tile 128x128, BK=32, 8 waves as 4(M)x2(N); wave tile 32x64.
// Tiles stream global->LDS via GLOBAL_LOAD_ASYNC_TO_LDS_B128, 4 buffers,
// 2 tiles in flight (ASYNCcnt-pipelined), one barrier per K-step.
// ---------------------------------------------------------------------------
__global__ __launch_bounds__(256) void enc_gemm(const unsigned short* __restrict__ xb,
                                                const unsigned short* __restrict__ Wb,
                                                const float* __restrict__ benc,
                                                unsigned short* __restrict__ preb) {
    __shared__ __attribute__((aligned(16))) unsigned short sA[4][128 * 32];  // 32 KB
    __shared__ __attribute__((aligned(16))) unsigned short sB[4][128 * 32];  // 32 KB

    const int t    = threadIdx.x;
    const int wave = t >> 5;
    const int lane = t & 31;
    const int col  = lane & 15;   // M (A) / N (B,C) index within a 16-wide tile
    const int half = lane >> 4;   // which K-half of the fragment
    const int wm   = wave >> 1;   // 0..3 : row group (32 rows each)
    const int wn   = wave & 1;    // 0..1 : col group (64 cols each)
    const int m0 = blockIdx.y * 128;
    const int h0 = blockIdx.x * 128;

    v8f acc[8] = {};   // acc[i*4+j] : rows wm*32+i*16, cols wn*64+j*16

    // cooperative async tile copy: 1024 x 16B per tile, 4 per thread
    const int r0 = t >> 2, c0 = t & 3;                 // row, 16B-chunk within row
    const unsigned aB0 = ((unsigned)(m0 + r0)      * DIM + c0 * 8) * 2;
    const unsigned aB1 = ((unsigned)(m0 + r0 + 64) * DIM + c0 * 8) * 2;
    const unsigned bB0 = ((unsigned)(h0 + r0)      * DIM + c0 * 8) * 2;
    const unsigned bB1 = ((unsigned)(h0 + r0 + 64) * DIM + c0 * 8) * 2;
    const unsigned ldsA = ldsOff(&sA[0][0]);
    const unsigned ldsB = ldsOff(&sB[0][0]);

    auto issueTile = [&](int kt) {
        const unsigned buf = (unsigned)(kt & 3) * 8192;
        const unsigned kb  = (unsigned)kt * 64;        // 32 bf16 = 64 bytes
        asyncCp16(ldsA + buf + t * 16,         aB0 + kb, xb);
        asyncCp16(ldsA + buf + (t + 256) * 16, aB1 + kb, xb);
        asyncCp16(ldsB + buf + t * 16,         bB0 + kb, Wb);
        asyncCp16(ldsB + buf + (t + 256) * 16, bB1 + kb, Wb);
    };

    const int KT = DIM / 32;   // 24
    issueTile(0);
    issueTile(1);

    for (int kt = 0; kt < KT; ++kt) {
        const int cur = kt & 3;
        if (kt + 2 < KT) {
            issueTile(kt + 2);
            asm volatile("s_wait_asynccnt 0x8" ::: "memory");   // tile kt landed
        } else if (kt + 1 < KT) {
            asm volatile("s_wait_asynccnt 0x4" ::: "memory");
        } else {
            asm volatile("s_wait_asynccnt 0x0" ::: "memory");
        }
        __syncthreads();   // also fences: buffer (kt+2)&3 was last read at kt-2

        union U { v16bf v; uint4 q[2]; };
        // A fragments (16x32): lane half -> K[0:8)+K[16:24) | K[8:16)+K[24:32)
        U af[2];
        #pragma unroll
        for (int i = 0; i < 2; ++i) {
            const unsigned short* arow = &sA[cur][(wm * 32 + i * 16 + col) * 32];
            af[i].q[0] = *(const uint4*)(arow + half * 8);
            af[i].q[1] = *(const uint4*)(arow + half * 8 + 16);
        }
        // B fragments (32x16): lane half -> K[0:8)+K[16:24) | K[8:16)+K[24:32) of its column
        U bfr[4];
        #pragma unroll
        for (int j = 0; j < 4; ++j) {
            const unsigned short* brow = &sB[cur][(wn * 64 + j * 16 + col) * 32];
            bfr[j].q[0] = *(const uint4*)(brow + half * 8);
            bfr[j].q[1] = *(const uint4*)(brow + half * 8 + 16);
        }
        #pragma unroll
        for (int i = 0; i < 2; ++i)
            #pragma unroll
            for (int j = 0; j < 4; ++j)
                acc[i * 4 + j] = __builtin_amdgcn_wmma_f32_16x16x32_bf16(
                    false, af[i].v, false, bfr[j].v, (short)0, acc[i * 4 + j], false, false);
    }

    // epilogue: C layout VGPR ii -> M = ii + 8*half, N = col; add bias, store bf16
    #pragma unroll
    for (int i = 0; i < 2; ++i) {
        #pragma unroll
        for (int j = 0; j < 4; ++j) {
            const int h = h0 + wn * 64 + j * 16 + col;
            const float bias = benc[h];
            #pragma unroll
            for (int ii = 0; ii < 8; ++ii) {
                const int m = m0 + wm * 32 + i * 16 + half * 8 + ii;
                preb[(size_t)m * HIDDEN + h] = f2bf(acc[i * 4 + j][ii] + bias);
            }
        }
    }
}

// ---------------------------------------------------------------------------
// Top-32 selection + sparse decode; one block per batch row.
// Incremental selection: only the stripe that lost its max rescans.
// ---------------------------------------------------------------------------
__global__ __launch_bounds__(256) void topk_decode(const unsigned short* __restrict__ preb,
                                                   const float* __restrict__ Wdec,
                                                   const float* __restrict__ bdec,
                                                   float* __restrict__ out) {
    __shared__ __attribute__((aligned(16))) unsigned short sPre[HIDDEN]; // 48 KB
    __shared__ float sVal[256], sRVal[256];
    __shared__ int   sIdx[256], sRIdx[256];
    __shared__ float sTopV[TOPK];
    __shared__ int   sTopI[TOPK];

    const int t = threadIdx.x;
    const int b = blockIdx.x;
    const unsigned short* row = preb + (size_t)b * HIDDEN;

    for (int i = t; i < HIDDEN / 8; i += 256)
        ((uint4*)sPre)[i] = ((const uint4*)row)[i];
    __syncthreads();

    // initial per-thread stripe maxima (stripe: i = t, t+256, ...)
    {
        float mv = -3.4e38f; int mi = t;
        for (int i = t; i < HIDDEN; i += 256) {
            const float v = bf2f(sPre[i]);
            if (v > mv) { mv = v; mi = i; }
        }
        sVal[t] = mv; sIdx[t] = mi;
    }
    __syncthreads();

    for (int it = 0; it < TOPK; ++it) {
        sRVal[t] = sVal[t]; sRIdx[t] = sIdx[t];
        __syncthreads();
        for (int s = 128; s > 0; s >>= 1) {
            if (t < s && sRVal[t + s] > sRVal[t]) { sRVal[t] = sRVal[t + s]; sRIdx[t] = sRIdx[t + s]; }
            __syncthreads();
        }
        const int   win = sRIdx[0];
        const float wv  = sRVal[0];
        if (t == 0) { sTopV[it] = wv; sTopI[it] = win; sPre[win] = 0xFF80; } // bf16 -inf
        __syncthreads();
        if (t == (win & 255)) {          // owner stripe rescans
            float mv = -3.4e38f; int mi = t;
            for (int i = t; i < HIDDEN; i += 256) {
                const float v = bf2f(sPre[i]);
                if (v > mv) { mv = v; mi = i; }
            }
            sVal[t] = mv; sIdx[t] = mi;
        }
        __syncthreads();
    }

    // out[b,d] = b_dec[d] + sum_j relu(topv_j) * W_dec[d, topi_j]
    for (int d = t; d < DIM; d += 256) {
        float acc = bdec[d];
        const float* wrow = Wdec + (size_t)d * HIDDEN;
        #pragma unroll
        for (int j = 0; j < TOPK; ++j) {
            const float v = sTopV[j];
            if (v > 0.f) acc += v * wrow[sTopI[j]];
        }
        out[(size_t)b * DIM + d] = acc;
    }
}

// ---------------------------------------------------------------------------
extern "C" void kernel_launch(void* const* d_in, const int* in_sizes, int n_in,
                              void* d_out, int out_size, void* d_ws, size_t ws_size,
                              hipStream_t stream) {
    const float* x    = (const float*)d_in[0];
    const float* Wenc = (const float*)d_in[1];
    const float* benc = (const float*)d_in[2];
    const float* Wdec = (const float*)d_in[3];
    const float* bdec = (const float*)d_in[4];
    float* out = (float*)d_out;

    char* ws = (char*)d_ws;
    unsigned short* Wb   = (unsigned short*)ws;                                      // 37.75 MB
    unsigned short* xb   = (unsigned short*)(ws + (size_t)HIDDEN * DIM * 2);         // 12.58 MB
    unsigned short* preb = (unsigned short*)(ws + (size_t)HIDDEN * DIM * 2
                                                + (size_t)BATCH * DIM * 2);          // 402.7 MB

    conv_wenc<<<2048, 256, 0, stream>>>(Wenc, Wb);
    conv_x<<<BATCH, 256, 0, stream>>>(x, bdec, xb);
    enc_gemm<<<dim3(HIDDEN / 128, BATCH / 128), 256, 0, stream>>>(xb, Wb, benc, preb);
    topk_decode<<<BATCH, 256, 0, stream>>>(preb, Wdec, bdec, out);
}
